// Intra_GCN_52329881534579
// MI455X (gfx1250) — matrix operand order repaired and checked by reading, not compiled, not run on hardware
//
#include <hip/hip_runtime.h>

#define N_NODES 10000
#define N_EDGES 160000
#define D_IN    512
#define D_OUT   512
#define N_SEG   256
#define LN_EPS  1e-5f
#define M_TILES (N_NODES / 16)   // 625, exact

typedef __attribute__((ext_vector_type(16))) __bf16 v16bf;
typedef __attribute__((ext_vector_type(8)))  float  v8f;

union Frag32B { v16bf v; uint4 q[2]; };

__device__ __forceinline__ unsigned short f2bf(float f) {
  union { float f; unsigned u; } x; x.f = f;
  unsigned r = x.u + 0x7FFFu + ((x.u >> 16) & 1u);   // round-to-nearest-even
  return (unsigned short)(r >> 16);
}

__device__ __forceinline__ float waveSum(float v) {
#pragma unroll
  for (int m = 16; m >= 1; m >>= 1) v += __shfl_xor(v, m, 32);
  return v;
}

__device__ __forceinline__ void atomAddF(float* p, float v) {
  unsafeAtomicAdd(p, v);   // global_atomic_add_f32 path
}

// ---------------- zero scratch ----------------
__global__ __launch_bounds__(256) void k_zero(float* __restrict__ p, int n) {
  int i = (blockIdx.x * 256 + threadIdx.x) * 4;
  if (i + 3 < n) {
    *(float4*)(p + i) = make_float4(0.f, 0.f, 0.f, 0.f);
  } else {
    for (int j = i; j < n; ++j) p[j] = 0.f;
  }
}

// ---------------- LN0: one wave per node row ----------------
__global__ __launch_bounds__(256) void k_ln0(const float* __restrict__ x,
                                             const float* __restrict__ w,
                                             const float* __restrict__ b,
                                             float* __restrict__ h,
                                             unsigned short* __restrict__ hbf) {
  int row  = blockIdx.x * 8 + (threadIdx.x >> 5);
  int lane = threadIdx.x & 31;
  if (row >= N_NODES) return;
  const float* xr = x + (size_t)row * D_IN + lane * 16;
  float vals[16];
#pragma unroll
  for (int q = 0; q < 4; ++q) {
    float4 t = ((const float4*)xr)[q];
    vals[4*q+0] = t.x; vals[4*q+1] = t.y; vals[4*q+2] = t.z; vals[4*q+3] = t.w;
  }
  float s = 0.f, ss = 0.f;
#pragma unroll
  for (int j = 0; j < 16; ++j) { s += vals[j]; ss += vals[j] * vals[j]; }
  s  = waveSum(s);
  ss = waveSum(ss);
  float mu  = s * (1.f / D_IN);
  float var = ss * (1.f / D_IN) - mu * mu;
  float rs  = rsqrtf(var + LN_EPS);
  int c0 = lane * 16;
  float*          hr = h   + (size_t)row * D_IN + c0;
  unsigned short* hb = hbf + (size_t)row * D_IN + c0;
#pragma unroll
  for (int j = 0; j < 16; ++j) {
    float y = (vals[j] - mu) * rs * w[c0 + j] + b[c0 + j];
    hr[j] = y;
    hb[j] = f2bf(y);
  }
}

// ---------------- weights f32 -> bf16 ----------------
__global__ __launch_bounds__(256) void k_wcvt(const float* __restrict__ Wn,
                                              const float* __restrict__ Wr,
                                              unsigned short* __restrict__ WnB,
                                              unsigned short* __restrict__ WrB) {
  int i = blockIdx.x * 256 + threadIdx.x;    // grid sized exactly to 512*512
  WnB[i] = f2bf(Wn[i]);
  WrB[i] = f2bf(Wr[i]);
}

// ---------------- edge scatter: one wave per edge ----------------
__global__ __launch_bounds__(256) void k_scatter(const int* __restrict__ ei,
                                                 const float* __restrict__ h,
                                                 float* __restrict__ agg,
                                                 float* __restrict__ deg) {
  int e    = blockIdx.x * 8 + (threadIdx.x >> 5);
  int lane = threadIdx.x & 31;
  if (e >= N_EDGES) return;
  int src = ei[e];
  int dst = ei[N_EDGES + e];
  const float* hr = h   + (size_t)src * D_IN + lane * 16;
  float*       ar = agg + (size_t)dst * D_IN + lane * 16;
  float vals[16];
#pragma unroll
  for (int q = 0; q < 4; ++q) {
    float4 t = ((const float4*)hr)[q];
    vals[4*q+0] = t.x; vals[4*q+1] = t.y; vals[4*q+2] = t.z; vals[4*q+3] = t.w;
  }
#pragma unroll
  for (int j = 0; j < 16; ++j) atomAddF(ar + j, vals[j]);
  if (lane == 0) atomAddF(deg + dst, 1.f);
}

// ---------------- mean = agg/deg -> bf16 ----------------
__global__ __launch_bounds__(256) void k_mean(const float* __restrict__ agg,
                                              const float* __restrict__ deg,
                                              unsigned short* __restrict__ mbf) {
  int i = blockIdx.x * 256 + threadIdx.x;    // grid exact: N_NODES*D_IN
  float d = deg[i >> 9];
  d = d > 1.f ? d : 1.f;
  mbf[i] = f2bf(agg[i] / d);
}

// ---------------- fused dual GEMM with bf16 WMMA ----------------
// out_pre = relu( mean @ Wn^T + h @ Wr^T + bias ), M=10000, N=512, K=512+512
// 8 waves/block; wave computes a 16(M) x 64(N) tile.
__global__ __launch_bounds__(256) void k_gemm(const unsigned short* __restrict__ mean_bf,
                                              const unsigned short* __restrict__ h_bf,
                                              const unsigned short* __restrict__ WnB,
                                              const unsigned short* __restrict__ WrB,
                                              const float* __restrict__ bias,
                                              float* __restrict__ out_pre) {
  const int wave  = threadIdx.x >> 5;
  const int lane  = threadIdx.x & 31;
  const int mtile = blockIdx.y * 8 + wave;     // wave-uniform
  if (mtile >= M_TILES) return;                // EXEC stays all-ones per wave
  const int m_base = mtile * 16;
  const int n_base = blockIdx.x * 64;          // gridDim.x = 8 -> N=512
  const int lm = lane & 15;                    // row/col within 16
  const int lh = lane >> 4;                    // half-wave select

  v8f acc[4] = {};

  const unsigned short* Aptr[2] = { mean_bf, h_bf };
  const unsigned short* Bptr[2] = { WnB, WrB };

#pragma unroll
  for (int ph = 0; ph < 2; ++ph) {
    const unsigned short* A = Aptr[ph] + (size_t)(m_base + lm) * D_IN;
    const unsigned short* B = Bptr[ph];
    for (int k0 = 0; k0 < D_IN; k0 += 32) {
      // A fragment: 16x32 bf16, lane holds row m_base+lm,
      // K = k0 + lh*16 + {0..15}
      Frag32B a;
      a.q[0] = *(const uint4*)(A + k0 + lh * 16);
      a.q[1] = *(const uint4*)(A + k0 + lh * 16 + 8);
#pragma unroll
      for (int nt = 0; nt < 4; ++nt) {
        // B fragment: 32x16 bf16, lane holds col n_base+nt*16+lm,
        // K = k0 + lh*16 + {0..15}  -> contiguous 32B from W[n][k] row-major
        const unsigned short* Bp = B + (size_t)(n_base + nt * 16 + lm) * D_IN + k0 + lh * 16;
        Frag32B bfr;
        bfr.q[0] = ((const uint4*)Bp)[0];
        bfr.q[1] = ((const uint4*)Bp)[1];
        acc[nt] = __builtin_amdgcn_wmma_f32_16x16x32_bf16(
            false, a.v, false, bfr.v, (short)0, acc[nt], false, false);
      }
    }
  }

  // epilogue: bias + relu, C layout: VGPR v -> row v (lanes 0-15) / v+8 (lanes 16-31)
#pragma unroll
  for (int nt = 0; nt < 4; ++nt) {
    const int col = n_base + nt * 16 + lm;
    const float bs = bias[col];
#pragma unroll
    for (int v = 0; v < 8; ++v) {
      const int row = m_base + v + lh * 8;
      float val = acc[nt][v] + bs;
      out_pre[(size_t)row * D_OUT + col] = val > 0.f ? val : 0.f;
    }
  }
}

// ---------------- LN1 + segment pooling: one wave per node row ----------------
__global__ __launch_bounds__(256) void k_ln1pool(const float* __restrict__ op,
                                                 const int* __restrict__ batch,
                                                 const float* __restrict__ w,
                                                 const float* __restrict__ b,
                                                 float* __restrict__ seg,
                                                 float* __restrict__ cnt) {
  int row  = blockIdx.x * 8 + (threadIdx.x >> 5);
  int lane = threadIdx.x & 31;
  if (row >= N_NODES) return;
  const float* r = op + (size_t)row * D_OUT + lane * 16;
  float vals[16];
#pragma unroll
  for (int q = 0; q < 4; ++q) {
    float4 t = ((const float4*)r)[q];
    vals[4*q+0] = t.x; vals[4*q+1] = t.y; vals[4*q+2] = t.z; vals[4*q+3] = t.w;
  }
  float s = 0.f, ss = 0.f;
#pragma unroll
  for (int j = 0; j < 16; ++j) { s += vals[j]; ss += vals[j] * vals[j]; }
  s  = waveSum(s);
  ss = waveSum(ss);
  float mu  = s * (1.f / D_OUT);
  float var = ss * (1.f / D_OUT) - mu * mu;
  float rs  = rsqrtf(var + LN_EPS);
  int sidx = batch[row];
  int c0   = lane * 16;
  float* dp = seg + (size_t)sidx * D_OUT + c0;
#pragma unroll
  for (int j = 0; j < 16; ++j)
    atomAddF(dp + j, (vals[j] - mu) * rs * w[c0 + j] + b[c0 + j]);
  if (lane == 0) atomAddF(cnt + sidx, 1.f);
}

// ---------------- finalize: seg mean ----------------
__global__ __launch_bounds__(256) void k_final(const float* __restrict__ seg,
                                               const float* __restrict__ cnt,
                                               float* __restrict__ out) {
  int i = blockIdx.x * 256 + threadIdx.x;   // grid exact: N_SEG*D_OUT
  float c = cnt[i >> 9];
  c = c > 1.f ? c : 1.f;
  out[i] = seg[i] / c;
}

// ---------------- workspace layout (bytes, 256-aligned) ----------------
#define OFF_H      0u          // f32   [10000][512]  20,480,000
#define OFF_HBF    20480000u   // bf16  [10000][512]  10,240,000
#define OFF_MBF    30720000u   // bf16  [10000][512]  10,240,000
#define OFF_WN     40960000u   // bf16  [512][512]       524,288
#define OFF_WR     41484288u   // bf16  [512][512]       524,288
#define OFF_AGG    42008576u   // f32   [10000][512]  20,480,000  (reused as out_pre)
#define OFF_DEG    62488576u   // f32   [10000]           40,960 (padded)
#define OFF_SEG    62529536u   // f32   [256][512]       524,288
#define OFF_CNT    63053824u   // f32   [256]              4,096 (padded)
#define WS_END     63057920u

extern "C" void kernel_launch(void* const* d_in, const int* in_sizes, int n_in,
                              void* d_out, int out_size, void* d_ws, size_t ws_size,
                              hipStream_t stream) {
  (void)in_sizes; (void)n_in; (void)out_size; (void)ws_size;
  const float* x     = (const float*)d_in[0];
  const int*   ei    = (const int*)d_in[1];
  const int*   batch = (const int*)d_in[2];
  const float* ln0w  = (const float*)d_in[3];
  const float* ln0b  = (const float*)d_in[4];
  const float* Wn    = (const float*)d_in[5];
  const float* Wr    = (const float*)d_in[6];
  const float* bias  = (const float*)d_in[7];
  const float* ln1w  = (const float*)d_in[8];
  const float* ln1b  = (const float*)d_in[9];

  char* ws = (char*)d_ws;
  float*          h      = (float*)(ws + OFF_H);
  unsigned short* hbf    = (unsigned short*)(ws + OFF_HBF);
  unsigned short* mbf    = (unsigned short*)(ws + OFF_MBF);
  unsigned short* WnB    = (unsigned short*)(ws + OFF_WN);
  unsigned short* WrB    = (unsigned short*)(ws + OFF_WR);
  float*          agg    = (float*)(ws + OFF_AGG);
  float*          outpre = agg;            // agg is dead after k_mean
  float*          deg    = (float*)(ws + OFF_DEG);
  float*          seg    = (float*)(ws + OFF_SEG);
  float*          cnt    = (float*)(ws + OFF_CNT);

  // zero agg..cnt span in one pass
  const int zn = (int)((WS_END - OFF_AGG) / 4);
  k_zero<<<(zn + 1023) / 1024, 256, 0, stream>>>(agg, zn);

  k_ln0<<<(N_NODES + 7) / 8, 256, 0, stream>>>(x, ln0w, ln0b, h, hbf);
  k_wcvt<<<(D_OUT * D_IN) / 256, 256, 0, stream>>>(Wn, Wr, WnB, WrB);
  k_scatter<<<(N_EDGES + 7) / 8, 256, 0, stream>>>(ei, h, agg, deg);
  k_mean<<<(N_NODES * D_IN) / 256, 256, 0, stream>>>(agg, deg, mbf);
  k_gemm<<<dim3(8, (M_TILES + 7) / 8), 256, 0, stream>>>(mbf, hbf, WnB, WrB, bias, outpre);
  k_ln1pool<<<(N_NODES + 7) / 8, 256, 0, stream>>>(outpre, batch, ln1w, ln1b, seg, cnt);
  k_final<<<(N_SEG * D_OUT) / 256, 256, 0, stream>>>(seg, cnt, (float*)d_out);
}
